// StructureModule_44573170598817
// MI455X (gfx1250) — compile-verified, matrix operand order checked by reading, not run on hardware
//
#include <hip/hip_runtime.h>
#include <hip/hip_bf16.h>
#include <math.h>

// ---------------- model dims ----------------
#define LSEQ   512
#define CS_D   256
#define CZ_D   128
#define H_D    8
#define CH_D   32
#define PQ_D   4
#define PV_D   8
#define NBLK   8
#define PROJW  1152   // 256 q | 256 k | 256 v | 96 qp | 96 kp | 192 vp
#define CAT_D  1536   // 256 o | 192 optl | 64 onrm | 1024 opair
#define AFRK   64     // padded K for fused logits GEMM (32 qk + 12 pts + 2 norm terms + pad)

typedef __attribute__((ext_vector_type(16))) _Float16 v16h;
typedef __attribute__((ext_vector_type(8)))  float    v8f;

#define W_L 0.5773502691896258f   // sqrt(1/3)
#define W_C 0.2357022603955158f   // sqrt(2/(9*PQ)) = sqrt(1/18)

__device__ __forceinline__ v8f wmma_fp16(v16h a, v16h b, v8f c) {
  // v_wmma_f32_16x16x32_f16  (neg_a, A, neg_b, B, c_mod, C, reuse_a, reuse_b)
  return __builtin_amdgcn_wmma_f32_16x16x32_f16(false, a, false, b, (short)0, c,
                                                false, false);
}

__device__ __forceinline__ float wave_sum(float v) {
#pragma unroll
  for (int off = 16; off > 0; off >>= 1) v += __shfl_xor(v, off, 32);
  return v;
}
__device__ __forceinline__ float wave_max(float v) {
#pragma unroll
  for (int off = 16; off > 0; off >>= 1) v = fmaxf(v, __shfl_xor(v, off, 32));
  return v;
}

// ---------------- WMMA fragment-resident B layout ----------------
// A 32x16 (KxN) B tile is stored as 32 lanes x 16 contiguous halfs, holding
// exactly what v_wmma_f32_16x16x32_f16 expects in each lane's 8 VGPRs:
//   lane = n' + 16*((k'>>3)&1),  elem = (k'&7) + 8*(k'>>4)
// Tiles are stored [kt][nt] row-major, 512 halfs (1 KB) per tile.
__device__ __forceinline__ long frag_off(int k, int col, int NT) {
  int kt = k >> 5, kp = k & 31;
  int nt = col >> 4, np = col & 15;
  int lane = np + (((kp >> 3) & 1) << 4);
  int el   = (kp & 7) + ((kp >> 4) << 3);
  return ((long)(kt * NT + nt) << 9) + (lane << 4) + el;
}

// B fragment load from packed layout: 32 contiguous bytes per lane (2x b128)
__device__ __forceinline__ v16h load_b_frag_packed(const _Float16* Wp, int NT,
                                                   int kt, int ntile) {
  int lane = threadIdx.x & 31;
  const _Float16* p = Wp + ((long)(kt * NT + ntile) << 9) + (lane << 4);
  v16h b;
#pragma unroll
  for (int i = 0; i < 16; ++i) b[i] = p[i];
  return b;
}

// A fragment (16x32, f16, row-major A, leading dim lda). No per-lane guard:
// rowMask lets short-M operands (opair, M=8) duplicate rows; the duplicated
// D rows are discarded in the epilogue, so this is branch/exec-mask free.
__device__ __forceinline__ v16h load_a_frag(const _Float16* A, long lda,
                                            int row0, int k0, int rowMask) {
  int lane = threadIdx.x & 31;
  int r  = row0 + ((lane & 15) & rowMask);
  int kb = k0 + ((lane >> 4) << 3);
  const _Float16* p = A + (long)r * lda + kb;
  v16h a;
#pragma unroll
  for (int i = 0; i < 8; ++i) { a[i] = p[i]; a[8 + i] = p[16 + i]; }
  return a;
}

// ---------------- generic batched WMMA GEMM: C = A(f16,MxK) @ Wfrag(KxN) + bias ---------------
// 2x2 register tiling per wave: two A fragments x two B fragments -> 4 WMMA
// chains per K step (2 b128 loads per WMMA). All call sites have M % 32 == 0
// and an even number of n-tiles.
__global__ void gemm_f16_kernel(const _Float16* A, long lda, long sA,
                                const _Float16* Wp, int NT, long sW,
                                const float* bias,
                                float* C, long ldc, long sC,
                                _Float16* Ch, long ldch, long sCh,
                                int M, int N, int K, int relu) {
  int bz = blockIdx.z;
  A  += (long)bz * sA;
  Wp += (long)bz * sW;
  int wv = threadIdx.x >> 5, lane = threadIdx.x & 31;
  int pt = blockIdx.x * (blockDim.x >> 5) + wv;     // pair-of-n-tiles index
  int npairs = ((N + 15) >> 4) >> 1;
  if (pt >= npairs) return;                          // wave-uniform
  int nt0  = pt << 1;
  int row0 = blockIdx.y << 5;                        // 32 rows per wave
  v8f acc[2][2] = {};
  for (int kt = 0; kt < (K >> 5); ++kt) {
    int k0 = kt << 5;
    v16h a0 = load_a_frag(A, lda, row0,      k0, 15);
    v16h a1 = load_a_frag(A, lda, row0 + 16, k0, 15);
    v16h b0 = load_b_frag_packed(Wp, NT, kt, nt0);
    v16h b1 = load_b_frag_packed(Wp, NT, kt, nt0 + 1);
    acc[0][0] = wmma_fp16(a0, b0, acc[0][0]);
    acc[0][1] = wmma_fp16(a0, b1, acc[0][1]);
    acc[1][0] = wmma_fp16(a1, b0, acc[1][0]);
    acc[1][1] = wmma_fp16(a1, b1, acc[1][1]);
  }
#pragma unroll
  for (int mt = 0; mt < 2; ++mt) {
#pragma unroll
    for (int tl = 0; tl < 2; ++tl) {
      int n = ((nt0 + tl) << 4) + (lane & 15);
      if (n >= N) continue;
      float bv = bias ? bias[n] : 0.0f;
#pragma unroll
      for (int r = 0; r < 8; ++r) {
        int m = row0 + (mt << 4) + r + ((lane >> 4) << 3);
        float v = acc[mt][tl][r] + bv;
        if (relu) v = fmaxf(v, 0.0f);
        if (C)  C[(long)bz * sC  + (long)m * ldc  + n] = v;
        if (Ch) Ch[(long)bz * sCh + (long)m * ldch + n] = (_Float16)v;
      }
    }
  }
}

// ---------------- fused IPA logits: per-head K=64 WMMA GEMM + pair-bias/mask epilogue ---------
// Afr: [H][L][64] f16 row-major (A operand); Bfr: per-head frag layout, NT=32, 2 k-tiles.
// 2x2 tiling: two i-tiles x two j-tiles per wave.
__global__ void logits_kernel(const _Float16* Afr, const _Float16* Bfr,
                              const float* pair_bias, const float* mask,
                              float* logits) {
  int wv = threadIdx.x >> 5, lane = threadIdx.x & 31;
  int jt0 = (blockIdx.x * (blockDim.x >> 5) + wv) << 1;
  int row0 = blockIdx.y << 5;
  int h = blockIdx.z;
  const _Float16* A  = Afr + (long)h * LSEQ * AFRK;
  const _Float16* Bh = Bfr + (long)h * 2 * 32 * 512;
  v8f acc[2][2] = {};
#pragma unroll
  for (int kt = 0; kt < 2; ++kt) {
    int k0 = kt << 5;
    v16h a0 = load_a_frag(A, AFRK, row0,      k0, 15);
    v16h a1 = load_a_frag(A, AFRK, row0 + 16, k0, 15);
    v16h b0 = load_b_frag_packed(Bh, 32, kt, jt0);
    v16h b1 = load_b_frag_packed(Bh, 32, kt, jt0 + 1);
    acc[0][0] = wmma_fp16(a0, b0, acc[0][0]);
    acc[0][1] = wmma_fp16(a0, b1, acc[0][1]);
    acc[1][0] = wmma_fp16(a1, b0, acc[1][0]);
    acc[1][1] = wmma_fp16(a1, b1, acc[1][1]);
  }
#pragma unroll
  for (int mt = 0; mt < 2; ++mt) {
#pragma unroll
    for (int tl = 0; tl < 2; ++tl) {
      int j = ((jt0 + tl) << 4) + (lane & 15);
      float mb = (mask[j] - 1.0f) * 1.0e9f;
#pragma unroll
      for (int r = 0; r < 8; ++r) {
        int i = row0 + (mt << 4) + r + ((lane >> 4) << 3);
        long o = ((long)h * LSEQ + i) * LSEQ + j;
        logits[o] = acc[mt][tl][r] + W_L * pair_bias[o] + mb;
      }
    }
  }
}

// ---------------- opair: per-i batched (8 x 512) @ (512 x 128) WMMA, writes f16 into cat ------
// zn_p: per-i frag layout, NT=8, 16 k-tiles. rowMask=7 duplicates A rows 8..15 (discarded).
__global__ void opair_kernel(const _Float16* a_h, const _Float16* zn_p,
                             _Float16* cat_h) {
  int lane = threadIdx.x & 31;
  int i    = blockIdx.y;
  int nt0  = blockIdx.x << 1;
  const _Float16* A  = a_h + (long)i * LSEQ;             // A[h][j], stride L*L per head
  const _Float16* Zb = zn_p + (long)i * 16 * 8 * 512;
  v8f acc0 = {}, acc1 = {};
#pragma unroll 4
  for (int kt = 0; kt < 16; ++kt) {
    v16h af = load_a_frag(A, (long)LSEQ * LSEQ, 0, kt << 5, 7);
    v16h b0 = load_b_frag_packed(Zb, 8, kt, nt0);
    v16h b1 = load_b_frag_packed(Zb, 8, kt, nt0 + 1);
    acc0 = wmma_fp16(af, b0, acc0);
    acc1 = wmma_fp16(af, b1, acc1);
  }
#pragma unroll
  for (int tile = 0; tile < 2; ++tile) {
    v8f acc = tile ? acc1 : acc0;
    int n = ((nt0 + tile) << 4) + (lane & 15);
#pragma unroll
    for (int r = 0; r < 8; ++r) {
      int h = r + ((lane >> 4) << 3);
      if (h < H_D)
        cat_h[(long)i * CAT_D + 512 + h * CZ_D + n] = (_Float16)acc[r];
    }
  }
}

// ---------------- LayerNorm kernels (wave-per-row) ----------------
__global__ void ln256_kernel(const float* x, const float* g, const float* b,
                             _Float16* outh) {
  int row = blockIdx.x * (blockDim.x >> 5) + (threadIdx.x >> 5);
  if (row >= LSEQ) return;
  int lane = threadIdx.x & 31;
  const float* xr = x + (long)row * CS_D;
  float v[8]; float s = 0.f;
#pragma unroll
  for (int i = 0; i < 8; ++i) { v[i] = xr[lane * 8 + i]; s += v[i]; }
  float mu = wave_sum(s) * (1.0f / CS_D);
  float var = 0.f;
#pragma unroll
  for (int i = 0; i < 8; ++i) { float d = v[i] - mu; var += d * d; }
  float inv = rsqrtf(wave_sum(var) * (1.0f / CS_D) + 1e-5f);
#pragma unroll
  for (int i = 0; i < 8; ++i) {
    int ch = lane * 8 + i;
    outh[(long)row * CS_D + ch] = (_Float16)((v[i] - mu) * inv * g[ch] + b[ch]);
  }
}

// LN of z (C=128) -> f16 in per-i frag layout (NT=8), fused with pair_bias = zn @ wpb
__global__ void ln_z_kernel(const float* z, const float* g, const float* b,
                            const float* wpb, _Float16* zn_p, float* pair_bias) {
  long row = (long)blockIdx.x * (blockDim.x >> 5) + (threadIdx.x >> 5); // i*L + j
  int lane = threadIdx.x & 31;
  int iseq = (int)(row >> 9), j = (int)(row & 511);
  const float* zr = z + row * CZ_D;
  float v[4]; float s = 0.f;
#pragma unroll
  for (int i = 0; i < 4; ++i) { v[i] = zr[lane * 4 + i]; s += v[i]; }
  float mu = wave_sum(s) * (1.0f / CZ_D);
  float var = 0.f;
#pragma unroll
  for (int i = 0; i < 4; ++i) { float d = v[i] - mu; var += d * d; }
  float inv = rsqrtf(wave_sum(var) * (1.0f / CZ_D) + 1e-5f);
  _Float16* zb = zn_p + (long)iseq * 16 * 8 * 512;
  float ph[H_D];
#pragma unroll
  for (int h = 0; h < H_D; ++h) ph[h] = 0.f;
#pragma unroll
  for (int i = 0; i < 4; ++i) {
    int ch = lane * 4 + i;
    float y = (v[i] - mu) * inv * g[ch] + b[ch];
    zb[frag_off(j, ch, 8)] = (_Float16)y;
#pragma unroll
    for (int h = 0; h < H_D; ++h) ph[h] += y * wpb[ch * H_D + h];
  }
#pragma unroll
  for (int h = 0; h < H_D; ++h) ph[h] = wave_sum(ph[h]);
  if (lane == 0) {
#pragma unroll
    for (int h = 0; h < H_D; ++h)
      pair_bias[(long)h * LSEQ * LSEQ + row] = ph[h];
  }
}

// residual + LN (in-place on sc), optional per-row mask on delta; also emits f16 copy
__global__ void resln_kernel(float* sc, const float* delta, const float* mask,
                             const float* g, const float* b, _Float16* sch) {
  int row = blockIdx.x * (blockDim.x >> 5) + (threadIdx.x >> 5);
  if (row >= LSEQ) return;
  int lane = threadIdx.x & 31;
  float m = mask ? mask[row] : 1.0f;
  float v[8]; float s = 0.f;
#pragma unroll
  for (int i = 0; i < 8; ++i) {
    int ch = lane * 8 + i;
    v[i] = sc[(long)row * CS_D + ch] + delta[(long)row * CS_D + ch] * m;
    s += v[i];
  }
  float mu = wave_sum(s) * (1.0f / CS_D);
  float var = 0.f;
#pragma unroll
  for (int i = 0; i < 8; ++i) { float d = v[i] - mu; var += d * d; }
  float inv = rsqrtf(wave_sum(var) * (1.0f / CS_D) + 1e-5f);
#pragma unroll
  for (int i = 0; i < 8; ++i) {
    int ch = lane * 8 + i;
    float y = (v[i] - mu) * inv * g[ch] + b[ch];
    sc[(long)row * CS_D + ch]  = y;
    sch[(long)row * CS_D + ch] = (_Float16)y;
  }
}

// ---------------- softmax over j (wave per (h,i) row), output f16 probs ----------------
__global__ void softmax_kernel(const float* logits, _Float16* a_h) {
  long row = (long)blockIdx.x * (blockDim.x >> 5) + (threadIdx.x >> 5); // H*L rows
  int lane = threadIdx.x & 31;
  const float* x = logits + row * LSEQ;
  float v[16]; float mx = -3.0e38f;
#pragma unroll
  for (int i = 0; i < 16; ++i) { v[i] = x[lane + i * 32]; mx = fmaxf(mx, v[i]); }
  mx = wave_max(mx);
  float s = 0.f;
#pragma unroll
  for (int i = 0; i < 16; ++i) { v[i] = __expf(v[i] - mx); s += v[i]; }
  float inv = 1.0f / wave_sum(s);
#pragma unroll
  for (int i = 0; i < 16; ++i)
    a_h[row * LSEQ + lane + i * 32] = (_Float16)(v[i] * inv);
}

// ---------------- build global points + fused-logits / value operand fragments ----------------
__global__ void prep_points_kernel(const float* proj, const float* R, const float* t,
                                   const float* head_w,
                                   _Float16* Afr, _Float16* Bfr,
                                   _Float16* vfrag, _Float16* vpgf) {
  int idx = blockIdx.x * blockDim.x + threadIdx.x;
  if (idx >= LSEQ * H_D) return;
  int l = idx >> 3, h = idx & 7;
  const float* pr = proj + (long)l * PROJW;
  const float* Rl = R + l * 9;
  float tx = t[l * 3 + 0], ty = t[l * 3 + 1], tz = t[l * 3 + 2];
  float gamma = log1pf(expf(head_w[h]));          // softplus
  float spt = W_L * W_C * gamma;
  const float qs = W_L * 0.17677669529663687f;    // wL / sqrt(CH)
  _Float16* Ar = Afr + ((long)h * LSEQ + l) * AFRK;
  _Float16* Bh = Bfr + (long)h * 2 * 32 * 512;    // per-head frag region, NT=32
  // scalar q (A row) and k (B frag, row index k'=channel, col=l)
#pragma unroll 8
  for (int c = 0; c < CH_D; ++c) {
    Ar[c] = (_Float16)(pr[h * CH_D + c] * qs);
    Bh[frag_off(c, l, 32)] = (_Float16)(pr[CS_D + h * CH_D + c]);
  }
  float qn = 0.f, kn = 0.f;
#pragma unroll
  for (int p = 0; p < PQ_D; ++p) {
    const float* qp = pr + 768 + h * 12 + p * 3;
    float gx = Rl[0]*qp[0] + Rl[1]*qp[1] + Rl[2]*qp[2] + tx;
    float gy = Rl[3]*qp[0] + Rl[4]*qp[1] + Rl[5]*qp[2] + ty;
    float gz = Rl[6]*qp[0] + Rl[7]*qp[1] + Rl[8]*qp[2] + tz;
    Ar[32 + p * 3 + 0] = (_Float16)(gx * spt);
    Ar[32 + p * 3 + 1] = (_Float16)(gy * spt);
    Ar[32 + p * 3 + 2] = (_Float16)(gz * spt);
    qn += gx * gx + gy * gy + gz * gz;
    const float* kp = pr + 864 + h * 12 + p * 3;
    float hx = Rl[0]*kp[0] + Rl[1]*kp[1] + Rl[2]*kp[2] + tx;
    float hy = Rl[3]*kp[0] + Rl[4]*kp[1] + Rl[5]*kp[2] + ty;
    float hz = Rl[6]*kp[0] + Rl[7]*kp[1] + Rl[8]*kp[2] + tz;
    Bh[frag_off(32 + p * 3 + 0, l, 32)] = (_Float16)hx;
    Bh[frag_off(32 + p * 3 + 1, l, 32)] = (_Float16)hy;
    Bh[frag_off(32 + p * 3 + 2, l, 32)] = (_Float16)hz;
    kn += hx * hx + hy * hy + hz * hz;
  }
  Ar[44] = (_Float16)qn;               // * B[44] = -0.5*spt
  Ar[45] = (_Float16)(-0.5f * spt);    // * B[45] = kn
  Bh[frag_off(44, l, 32)] = (_Float16)(-0.5f * spt);
  Bh[frag_off(45, l, 32)] = (_Float16)kn;
#pragma unroll
  for (int k = 46; k < AFRK; ++k) {
    Ar[k] = (_Float16)0.f;
    Bh[frag_off(k, l, 32)] = (_Float16)0.f;
  }
  // value channels -> frag layout (row index = l, cols h*32..h*32+31, NT=16)
#pragma unroll 8
  for (int c = 0; c < CH_D; ++c)
    vfrag[frag_off(l, h * CH_D + c, 16)] = (_Float16)pr[512 + h * CH_D + c];
  // v points in global frame -> frag layout, padded to 32 cols per head
#pragma unroll
  for (int p = 0; p < PV_D; ++p) {
    const float* vp = pr + 960 + h * 24 + p * 3;
    float gx = Rl[0]*vp[0] + Rl[1]*vp[1] + Rl[2]*vp[2] + tx;
    float gy = Rl[3]*vp[0] + Rl[4]*vp[1] + Rl[5]*vp[2] + ty;
    float gz = Rl[6]*vp[0] + Rl[7]*vp[1] + Rl[8]*vp[2] + tz;
    vpgf[frag_off(l, h * 32 + p * 3 + 0, 16)] = (_Float16)gx;
    vpgf[frag_off(l, h * 32 + p * 3 + 1, 16)] = (_Float16)gy;
    vpgf[frag_off(l, h * 32 + p * 3 + 2, 16)] = (_Float16)gz;
  }
#pragma unroll
  for (int e = 24; e < 32; ++e)
    vpgf[frag_off(l, h * 32 + e, 16)] = (_Float16)0.f;
}

// optl = R^T (optg - t), onrm; write into cat columns 256..511
__global__ void build_cat_kernel(const float* optg, const float* R, const float* t,
                                 _Float16* cat_h) {
  int idx = blockIdx.x * blockDim.x + threadIdx.x;  // l*64 + h*8 + p
  if (idx >= LSEQ * H_D * PV_D) return;
  int l = idx >> 6, hp = idx & 63, h = hp >> 3, p = hp & 7;
  const float* og = optg + (long)l * 256 + h * 32 + p * 3;
  const float* Rl = R + l * 9;
  float dx = og[0] - t[l * 3 + 0];
  float dy = og[1] - t[l * 3 + 1];
  float dz = og[2] - t[l * 3 + 2];
  float lx = Rl[0] * dx + Rl[3] * dy + Rl[6] * dz;
  float ly = Rl[1] * dx + Rl[4] * dy + Rl[7] * dz;
  float lz = Rl[2] * dx + Rl[5] * dy + Rl[8] * dz;
  _Float16* cr = cat_h + (long)l * CAT_D;
  cr[256 + hp * 3 + 0] = (_Float16)lx;
  cr[256 + hp * 3 + 1] = (_Float16)ly;
  cr[256 + hp * 3 + 2] = (_Float16)lz;
  cr[448 + hp] = (_Float16)sqrtf(lx * lx + ly * ly + lz * lz + 1e-8f);
}

// ---------------- backbone update ----------------
__global__ void upd_kernel(const float* sc, const float* wbb, const float* bbb,
                           float* upd) {
  int idx = blockIdx.x * blockDim.x + threadIdx.x;
  if (idx >= LSEQ * 6) return;
  int l = idx / 6, d = idx % 6;
  float s = bbb[d];
  for (int k = 0; k < CS_D; ++k) s += sc[(long)l * CS_D + k] * wbb[k * 6 + d];
  upd[idx] = s;
}

__global__ void frame_init_kernel(float* R, float* t) {
  int l = blockIdx.x * blockDim.x + threadIdx.x;
  if (l >= LSEQ) return;
  float* Rl = R + l * 9;
  Rl[0] = 1.f; Rl[1] = 0.f; Rl[2] = 0.f;
  Rl[3] = 0.f; Rl[4] = 1.f; Rl[5] = 0.f;
  Rl[6] = 0.f; Rl[7] = 0.f; Rl[8] = 1.f;
  t[l * 3 + 0] = 0.f; t[l * 3 + 1] = 0.f; t[l * 3 + 2] = 0.f;
}

__global__ void frame_update_kernel(const float* upd, const float* mask,
                                    float* R, float* t) {
  int l = blockIdx.x * blockDim.x + threadIdx.x;
  if (l >= LSEQ) return;
  const float* u = upd + l * 6;
  float bq = u[0], cq = u[1], dq = u[2];
  float inv = rsqrtf(1.f + bq * bq + cq * cq + dq * dq);
  float a = inv, b = bq * inv, c = cq * inv, d = dq * inv;
  float dR[9] = {
    a*a + b*b - c*c - d*d, 2.f*(b*c - a*d),       2.f*(b*d + a*c),
    2.f*(b*c + a*d),       a*a - b*b + c*c - d*d, 2.f*(c*d - a*b),
    2.f*(b*d - a*c),       2.f*(c*d + a*b),       a*a - b*b - c*c + d*d };
  float m = mask[l];
  float dt[3] = { u[3] * m, u[4] * m, u[5] * m };
  float* Rl = R + l * 9;
  float* tl = t + l * 3;
  float Ro[9];
#pragma unroll
  for (int i = 0; i < 9; ++i) Ro[i] = Rl[i];
#pragma unroll
  for (int i = 0; i < 3; ++i)
    tl[i] = Ro[i*3+0]*dt[0] + Ro[i*3+1]*dt[1] + Ro[i*3+2]*dt[2] + tl[i];
#pragma unroll
  for (int i = 0; i < 3; ++i)
#pragma unroll
    for (int k = 0; k < 3; ++k)
      Rl[i*3+k] = Ro[i*3+0]*dR[0*3+k] + Ro[i*3+1]*dR[1*3+k] + Ro[i*3+2]*dR[2*3+k];
}

// ---------------- weight packing (f32 row-major -> f16 WMMA-frag layout) ----------------
__global__ void pack_w_frag_kernel(const float* src, _Float16* dst, int K, int Nsrc,
                                   int NT, int colOff) {
  long idx = (long)blockIdx.x * blockDim.x + threadIdx.x;
  if (idx >= (long)K * Nsrc) return;
  int k = (int)(idx / Nsrc), n = (int)(idx % Nsrc);
  dst[frag_off(k, colOff + n, NT)] = (_Float16)src[idx];
}

__global__ void copyf_kernel(const float* s, float* d, long n) {
  long i = (long)blockIdx.x * blockDim.x + threadIdx.x;
  if (i < n) d[i] = s[i];
}

// ==================================================================================
extern "C" void kernel_launch(void* const* d_in, const int* in_sizes, int n_in,
                              void* d_out, int out_size, void* d_ws, size_t ws_size,
                              hipStream_t stream) {
  (void)in_sizes; (void)n_in; (void)out_size; (void)ws_size;
  const float* s_in   = (const float*)d_in[0];
  const float* z_in   = (const float*)d_in[1];
  const float* mask   = (const float*)d_in[2];
  const float* ln_s_g = (const float*)d_in[3];
  const float* ln_s_b = (const float*)d_in[4];
  const float* ln_z_g = (const float*)d_in[5];
  const float* ln_z_b = (const float*)d_in[6];
  const float* w_init = (const float*)d_in[7];
  const float* b_init = (const float*)d_in[8];
  const float* ln_ipa_g = (const float*)d_in[9];
  const float* ln_ipa_b = (const float*)d_in[10];
  const float* ln_tr_g  = (const float*)d_in[11];
  const float* ln_tr_b  = (const float*)d_in[12];
  const float* wq  = (const float*)d_in[13];
  const float* bq  = (const float*)d_in[14];
  const float* wk  = (const float*)d_in[15];
  const float* bk  = (const float*)d_in[16];
  const float* wv  = (const float*)d_in[17];
  const float* bv  = (const float*)d_in[18];
  const float* wqp = (const float*)d_in[19];
  const float* bqp = (const float*)d_in[20];
  const float* wkp = (const float*)d_in[21];
  const float* bkp = (const float*)d_in[22];
  const float* wvp = (const float*)d_in[23];
  const float* bvp = (const float*)d_in[24];
  const float* wpb = (const float*)d_in[25];
  const float* head_w = (const float*)d_in[26];
  const float* wo  = (const float*)d_in[27];
  const float* bo  = (const float*)d_in[28];
  const float* wt1 = (const float*)d_in[29];
  const float* bt1 = (const float*)d_in[30];
  const float* wt2 = (const float*)d_in[31];
  const float* bt2 = (const float*)d_in[32];
  const float* wt3 = (const float*)d_in[33];
  const float* bt3 = (const float*)d_in[34];
  const float* wbb = (const float*)d_in[35];
  const float* bbb = (const float*)d_in[36];
  float* out = (float*)d_out;

  const long LL = (long)LSEQ * LSEQ;
  size_t off = 0;
  auto carve = [&](size_t bytes) {
    size_t o = off;
    off += (bytes + 255) & ~(size_t)255;
    return o;
  };
  char* ws = (char*)d_ws;
  _Float16* zn_p   = (_Float16*)(ws + carve((size_t)LL * CZ_D * 2));      // 67 MB, L2-resident
  float*    pairb  = (float*)   (ws + carve((size_t)H_D * LL * 4));       // 8 MB
  float*    logits = (float*)   (ws + carve((size_t)H_D * LL * 4));       // 8 MB
  _Float16* a_h    = (_Float16*)(ws + carve((size_t)H_D * LL * 2));       // 4 MB
  _Float16* s0h    = (_Float16*)(ws + carve((size_t)LSEQ * CS_D * 2));
  float*    sc     = (float*)   (ws + carve((size_t)LSEQ * CS_D * 4));
  _Float16* sch    = (_Float16*)(ws + carve((size_t)LSEQ * CS_D * 2));
  _Float16* winit_f= (_Float16*)(ws + carve((size_t)CS_D * CS_D * 2));
  _Float16* projW_f= (_Float16*)(ws + carve((size_t)CS_D * PROJW * 2));
  float*    biasP  = (float*)   (ws + carve((size_t)PROJW * 4));
  _Float16* wo_f   = (_Float16*)(ws + carve((size_t)CAT_D * CS_D * 2));
  _Float16* wt1_f  = (_Float16*)(ws + carve((size_t)CS_D * CS_D * 2));
  _Float16* wt2_f  = (_Float16*)(ws + carve((size_t)CS_D * CS_D * 2));
  _Float16* wt3_f  = (_Float16*)(ws + carve((size_t)CS_D * CS_D * 2));
  float*    proj   = (float*)   (ws + carve((size_t)LSEQ * PROJW * 4));
  _Float16* Afr    = (_Float16*)(ws + carve((size_t)H_D * LSEQ * AFRK * 2));
  _Float16* Bfr    = (_Float16*)(ws + carve((size_t)H_D * AFRK * LSEQ * 2));
  _Float16* vfrag  = (_Float16*)(ws + carve((size_t)LSEQ * 256 * 2));
  _Float16* vpgf   = (_Float16*)(ws + carve((size_t)LSEQ * 256 * 2));
  float*    optg   = (float*)   (ws + carve((size_t)LSEQ * 256 * 4));
  _Float16* cat_h  = (_Float16*)(ws + carve((size_t)LSEQ * CAT_D * 2));
  float*    ipa_up = (float*)   (ws + carve((size_t)LSEQ * CS_D * 4));
  _Float16* tr1_h  = (_Float16*)(ws + carve((size_t)LSEQ * CS_D * 2));
  _Float16* tr2_h  = (_Float16*)(ws + carve((size_t)LSEQ * CS_D * 2));
  float*    trf    = (float*)   (ws + carve((size_t)LSEQ * CS_D * 4));
  float*    updb   = (float*)   (ws + carve((size_t)LSEQ * 6 * 4));
  float*    Rb     = (float*)   (ws + carve((size_t)LSEQ * 9 * 4));
  float*    tb     = (float*)   (ws + carve((size_t)LSEQ * 3 * 4));

  auto gemm = [&](const _Float16* A, long lda, long sA,
                  const _Float16* Wp, int NT, long sW, const float* bias,
                  float* C, long ldc, long sC,
                  _Float16* Ch, long ldch, long sCh,
                  int M, int N, int K, int relu, int batch) {
    int npairs = ((N + 15) >> 4) >> 1, mt32 = (M + 31) >> 5;
    dim3 grid((npairs + 3) / 4, mt32, batch);
    gemm_f16_kernel<<<grid, 128, 0, stream>>>(A, lda, sA, Wp, NT, sW, bias,
                                              C, ldc, sC, Ch, ldch, sCh,
                                              M, N, K, relu);
  };
  auto pack = [&](const float* src, _Float16* dst, int K, int Nsrc, int NT, int co) {
    long n = (long)K * Nsrc;
    pack_w_frag_kernel<<<(unsigned)((n + 255) / 256), 256, 0, stream>>>(src, dst, K,
                                                                        Nsrc, NT, co);
  };
  auto copyf = [&](const float* srcp, float* dstp, long n) {
    copyf_kernel<<<(unsigned)((n + 255) / 256), 256, 0, stream>>>(srcp, dstp, n);
  };

  // ---- one-time setup: pack weights into WMMA-frag layout ----
  pack(w_init, winit_f, CS_D, CS_D, 16, 0);
  pack(wq,  projW_f, CS_D, 256, 72, 0);
  pack(wk,  projW_f, CS_D, 256, 72, 256);
  pack(wv,  projW_f, CS_D, 256, 72, 512);
  pack(wqp, projW_f, CS_D, 96,  72, 768);
  pack(wkp, projW_f, CS_D, 96,  72, 864);
  pack(wvp, projW_f, CS_D, 192, 72, 960);
  pack(wo,  wo_f,  CAT_D, CS_D, 16, 0);
  pack(wt1, wt1_f, CS_D, CS_D, 16, 0);
  pack(wt2, wt2_f, CS_D, CS_D, 16, 0);
  pack(wt3, wt3_f, CS_D, CS_D, 16, 0);
  copyf(bq,  biasP + 0,   256);
  copyf(bk,  biasP + 256, 256);
  copyf(bv,  biasP + 512, 256);
  copyf(bqp, biasP + 768, 96);
  copyf(bkp, biasP + 864, 96);
  copyf(bvp, biasP + 960, 192);

  ln256_kernel<<<64, 256, 0, stream>>>(s_in, ln_s_g, ln_s_b, s0h);
  ln_z_kernel<<<(unsigned)(LL / 8), 256, 0, stream>>>(z_in, ln_z_g, ln_z_b, wpb,
                                                      zn_p, pairb);
  frame_init_kernel<<<2, 256, 0, stream>>>(Rb, tb);
  // sc = LN(s) @ w_init + b_init
  gemm(s0h, CS_D, 0, winit_f, 16, 0, b_init,
       sc, CS_D, 0, sch, CS_D, 0, LSEQ, CS_D, CS_D, 0, 1);

  // ---- 8 IPA blocks ----
  for (int blk = 0; blk < NBLK; ++blk) {
    // fused projection: [q|k|v|qp|kp|vp] = sc @ Wproj + bias
    gemm(sch, CS_D, 0, projW_f, 72, 0, biasP,
         proj, PROJW, 0, nullptr, 0, 0, LSEQ, PROJW, CS_D, 0, 1);
    prep_points_kernel<<<16, 256, 0, stream>>>(proj, Rb, tb, head_w,
                                               Afr, Bfr, vfrag, vpgf);
    {
      dim3 grid(4, 16, H_D);   // 16 j-tile-pairs (4 waves/block), 16 i-tile-pairs, 8 heads
      logits_kernel<<<grid, 128, 0, stream>>>(Afr, Bfr, pairb, mask, logits);
    }
    softmax_kernel<<<512, 256, 0, stream>>>(logits, a_h);
    // o = a @ v  (per head, f16 straight into cat cols 0..255)
    gemm(a_h, LSEQ, LL, vfrag, 16, 1024, nullptr,
         nullptr, 0, 0, cat_h, CAT_D, 32, LSEQ, 32, LSEQ, 0, H_D);
    // optg = a @ vpg (per head, padded N, guard 24)
    gemm(a_h, LSEQ, LL, vpgf, 16, 1024, nullptr,
         optg, 256, 32, nullptr, 0, 0, LSEQ, 24, LSEQ, 0, H_D);
    {
      dim3 grid(CZ_D / 32, LSEQ);   // 4 c-tile-pairs per i
      opair_kernel<<<grid, 32, 0, stream>>>(a_h, zn_p, cat_h);
    }
    build_cat_kernel<<<128, 256, 0, stream>>>(optg, Rb, tb, cat_h);
    // ipa_up = cat @ wo + bo ; sc = LN(sc + ipa_up)
    gemm(cat_h, CAT_D, 0, wo_f, 16, 0, bo,
         ipa_up, CS_D, 0, nullptr, 0, 0, LSEQ, CS_D, CAT_D, 0, 1);
    resln_kernel<<<64, 256, 0, stream>>>(sc, ipa_up, nullptr,
                                         ln_ipa_g, ln_ipa_b, sch);
    // transition MLP
    gemm(sch, CS_D, 0, wt1_f, 16, 0, bt1,
         nullptr, 0, 0, tr1_h, CS_D, 0, LSEQ, CS_D, CS_D, 1, 1);
    gemm(tr1_h, CS_D, 0, wt2_f, 16, 0, bt2,
         nullptr, 0, 0, tr2_h, CS_D, 0, LSEQ, CS_D, CS_D, 1, 1);
    gemm(tr2_h, CS_D, 0, wt3_f, 16, 0, bt3,
         trf, CS_D, 0, nullptr, 0, 0, LSEQ, CS_D, CS_D, 0, 1);
    resln_kernel<<<64, 256, 0, stream>>>(sc, trf, mask,
                                         ln_tr_g, ln_tr_b, sch);
    // backbone frame update
    upd_kernel<<<12, 256, 0, stream>>>(sc, wbb, bbb, updb);
    frame_update_kernel<<<2, 256, 0, stream>>>(updb, mask, Rb, tb);
  }

  // outputs: (sc, R, t) flattened
  copyf(sc, out, (long)LSEQ * CS_D);
  copyf(Rb, out + (long)LSEQ * CS_D, (long)LSEQ * 9);
  copyf(tb, out + (long)LSEQ * CS_D + (long)LSEQ * 9, (long)LSEQ * 3);
}
